// TokenGTEmbedding_37838661878594
// MI455X (gfx1250) — matrix operand order repaired and checked by reading, not compiled, not run on hardware
//
#include <hip/hip_runtime.h>
#include <hip/hip_bf16.h>

typedef _Float16 f16;
typedef __attribute__((ext_vector_type(16))) _Float16 v16h;
typedef __attribute__((ext_vector_type(8)))  _Float16 v8h;
typedef __attribute__((ext_vector_type(8)))  float    v8f;

#define N_NODES 50000
#define N_EDGES 800000
#define F_DIMS  128
#define P_DIMS  64
#define H_DIMS  512

// workspace byte offsets (all 16B+ aligned)
#define OFF_WNT   0u          // f16 [512][128]  W[0:128]^T        (131072 B)
#define OFF_W2T   131072u     // f16 [512][64]   (W[128:192]+W[192:256])^T
#define OFF_WAT   196608u     // f16 [512][64]   W[128:192]^T
#define OFF_WBT   262144u     // f16 [512][64]   W[192:256]^T
#define OFF_BN    327680u     // f32 [512] node bias
#define OFF_BE    329728u     // f32 [512] edge bias
#define OFF_NF16  331776u     // f16 [N][128]
#define OFF_PE16  13131776u   // f16 [N][64]
// total ~19.6 MB

__global__ __launch_bounds__(256)
void pack_weights(const float* __restrict__ W, const float* __restrict__ b,
                  const float* __restrict__ ef, const float* __restrict__ tt,
                  f16* __restrict__ wnT, f16* __restrict__ w2T,
                  f16* __restrict__ waT, f16* __restrict__ wbT,
                  float* __restrict__ biasN, float* __restrict__ biasE) {
    int n = blockIdx.x * blockDim.x + threadIdx.x;
    if (n >= H_DIMS) return;
    float accN = b[n];
    float accE = b[n];
    for (int k = 0; k < 128; ++k) {
        float w0 = W[k * H_DIMS + n];            // node-feature / edge-feature block
        wnT[n * 128 + k] = (f16)w0;
        float w3 = W[(256 + k) * H_DIMS + n];    // token-type block
        accN += tt[k] * w3;                      // tt row 0 -> nodes
        accE += tt[128 + k] * w3 + ef[k] * w0;   // tt row 1 + edge_features -> edges
    }
    for (int k = 0; k < 64; ++k) {
        float wa = W[(128 + k) * H_DIMS + n];    // pe_src block
        float wb = W[(192 + k) * H_DIMS + n];    // pe_dst block
        waT[n * 64 + k] = (f16)wa;
        wbT[n * 64 + k] = (f16)wb;
        w2T[n * 64 + k] = (f16)(wa + wb);        // nodes use same pe twice
    }
    biasN[n] = accN;
    biasE[n] = accE;
}

__global__ __launch_bounds__(256)
void convert_f16(const float* __restrict__ nf, const float* __restrict__ pe,
                 f16* __restrict__ nf16, f16* __restrict__ pe16) {
    long i = (long)blockIdx.x * blockDim.x + threadIdx.x;
    const long nNF = (long)N_NODES * F_DIMS;
    const long nPE = (long)N_NODES * P_DIMS;
    if (i < nNF) {
        nf16[i] = (f16)nf[i];
    } else {
        long j = i - nNF;
        if (j < nPE) pe16[j] = (f16)pe[j];
    }
}

// ---- WMMA operand packing (wave32): contiguous k order, not the ISA table order ----
// A 16x32 f16: lane l -> row M = l%16, half = l/16.
//   halves j = K[kbase + 16*half + j]  (contiguous, matches B)
__device__ __forceinline__ v16h load_a(const f16* __restrict__ row, int kbase, int half) {
    union { v16h v; v8h h[2]; } u;
    u.h[0] = *(const v8h*)(row + kbase + 16 * half);
    u.h[1] = *(const v8h*)(row + kbase + 16 * half + 8);
    return u.v;
}
// B 32x16 f16 from a transposed [512][K] weight copy:
//   lane l -> col N = l%16, halves j = K[kbase + 16*half + j]  (32 contiguous bytes)
__device__ __forceinline__ v16h load_b(const f16* __restrict__ btRow, int kbase, int half) {
    const v8h* q = (const v8h*)(btRow + kbase + 16 * half);
    union { v16h v; v8h h[2]; } u;
    u.h[0] = q[0];
    u.h[1] = q[1];
    return u.v;
}
__device__ __forceinline__ v8f wmma_f16(v16h a, v16h b, v8f c) {
    return __builtin_amdgcn_wmma_f32_16x16x32_f16(false, a, false, b, (short)0, c, false, false);
}

// One wave: 16 rows x 64 cols. Block (8 waves): 16 rows x 512 cols.
__global__ __launch_bounds__(256)
void node_gemm(const f16* __restrict__ nf16, const f16* __restrict__ pe16,
               const f16* __restrict__ wnT, const f16* __restrict__ w2T,
               const float* __restrict__ biasN, float* __restrict__ out) {
    int lane = threadIdx.x & 31;
    int wave = threadIdx.x >> 5;
    int half = lane >> 4;
    int m    = lane & 15;
    int rowBase = blockIdx.x * 16;
    int colBase = wave * 64;

    const f16* aRowN = nf16 + (long)(rowBase + m) * F_DIMS;
    const f16* aRowP = pe16 + (long)(rowBase + m) * P_DIMS;

    v8f acc[4] = {};
    // phase 1: node features, K = 128
    #pragma unroll
    for (int kt = 0; kt < 4; ++kt) {
        v16h a = load_a(aRowN, kt * 32, half);
        #pragma unroll
        for (int c = 0; c < 4; ++c) {
            const f16* bt = wnT + (long)(colBase + 16 * c + m) * F_DIMS + kt * 32;
            acc[c] = wmma_f16(a, load_b(bt, 0, half), acc[c]);
        }
    }
    // phase 2: positional encoding with folded (Wpe1+Wpe2), K = 64
    #pragma unroll
    for (int kt = 0; kt < 2; ++kt) {
        v16h a = load_a(aRowP, kt * 32, half);
        #pragma unroll
        for (int c = 0; c < 4; ++c) {
            const f16* bt = w2T + (long)(colBase + 16 * c + m) * P_DIMS + kt * 32;
            acc[c] = wmma_f16(a, load_b(bt, 0, half), acc[c]);
        }
    }
    // D layout: lane -> col N = m, VGPR r -> row M = r + 8*half
    #pragma unroll
    for (int c = 0; c < 4; ++c) {
        int col = colBase + 16 * c + m;
        float bn = biasN[col];
        float* o = out + (long)rowBase * H_DIMS + col;
        #pragma unroll
        for (int r = 0; r < 8; ++r)
            o[(long)(r + 8 * half) * H_DIMS] = acc[c][r] + bn;
    }
}

// One wave: 16 edges x 64 cols; pe rows gathered per lane via edge_index.
__global__ __launch_bounds__(256)
void edge_gemm(const long long* __restrict__ eidx, const f16* __restrict__ pe16,
               const f16* __restrict__ waT, const f16* __restrict__ wbT,
               const float* __restrict__ biasE, float* __restrict__ out) {
    int lane = threadIdx.x & 31;
    int wave = threadIdx.x >> 5;
    int half = lane >> 4;
    int m    = lane & 15;
    long eBase  = (long)blockIdx.x * 16;
    int colBase = wave * 64;

    long e = eBase + m;
    int s = (int)eidx[e];
    int d = (int)eidx[(long)N_EDGES + e];
    const f16* aS = pe16 + (long)s * P_DIMS;
    const f16* aD = pe16 + (long)d * P_DIMS;

    v8f acc[4] = {};
    // pe[src] @ Wa, K = 64
    #pragma unroll
    for (int kt = 0; kt < 2; ++kt) {
        v16h a = load_a(aS, kt * 32, half);
        #pragma unroll
        for (int c = 0; c < 4; ++c) {
            const f16* bt = waT + (long)(colBase + 16 * c + m) * P_DIMS + kt * 32;
            acc[c] = wmma_f16(a, load_b(bt, 0, half), acc[c]);
        }
    }
    // pe[dst] @ Wb, K = 64, accumulated into same tile
    #pragma unroll
    for (int kt = 0; kt < 2; ++kt) {
        v16h a = load_a(aD, kt * 32, half);
        #pragma unroll
        for (int c = 0; c < 4; ++c) {
            const f16* bt = wbT + (long)(colBase + 16 * c + m) * P_DIMS + kt * 32;
            acc[c] = wmma_f16(a, load_b(bt, 0, half), acc[c]);
        }
    }
    long rowBase = (long)N_NODES + eBase;
    #pragma unroll
    for (int c = 0; c < 4; ++c) {
        int col = colBase + 16 * c + m;
        float be = biasE[col];
        float* o = out + rowBase * H_DIMS + col;
        #pragma unroll
        for (int r = 0; r < 8; ++r)
            o[(long)(r + 8 * half) * H_DIMS] = acc[c][r] + be;
    }
}

extern "C" void kernel_launch(void* const* d_in, const int* in_sizes, int n_in,
                              void* d_out, int out_size, void* d_ws, size_t ws_size,
                              hipStream_t stream) {
    const float*     nf   = (const float*)d_in[0];   // [N,128]
    const float*     pe   = (const float*)d_in[1];   // [N,64]
    const float*     ef   = (const float*)d_in[2];   // [1,128]
    const float*     tt   = (const float*)d_in[3];   // [2,128]
    const float*     W    = (const float*)d_in[4];   // [384,512]
    const float*     b    = (const float*)d_in[5];   // [512]
    const long long* eidx = (const long long*)d_in[6]; // [2,E] int64
    float* out = (float*)d_out;

    char* ws = (char*)d_ws;
    f16*   wnT   = (f16*)  (ws + OFF_WNT);
    f16*   w2T   = (f16*)  (ws + OFF_W2T);
    f16*   waT   = (f16*)  (ws + OFF_WAT);
    f16*   wbT   = (f16*)  (ws + OFF_WBT);
    float* biasN = (float*)(ws + OFF_BN);
    float* biasE = (float*)(ws + OFF_BE);
    f16*   nf16  = (f16*)  (ws + OFF_NF16);
    f16*   pe16  = (f16*)  (ws + OFF_PE16);

    // 1) fold constants into biases, transpose+convert weights (stream-ordered deps)
    pack_weights<<<2, 256, 0, stream>>>(W, b, ef, tt, wnT, w2T, waT, wbT, biasN, biasE);

    // 2) f32 -> f16 conversion of node features + positional encodings
    long totalConv = (long)N_NODES * (F_DIMS + P_DIMS);   // 9.6M elements
    convert_f16<<<(int)((totalConv + 255) / 256), 256, 0, stream>>>(nf, pe, nf16, pe16);

    // 3) node rows: [50000,192] @ folded weights, 16 rows per block
    node_gemm<<<N_NODES / 16, 256, 0, stream>>>(nf16, pe16, wnT, w2T, biasN, out);

    // 4) edge rows: gathered pe @ Wa/Wb, 16 edges per block
    edge_gemm<<<N_EDGES / 16, 256, 0, stream>>>(eidx, pe16, waT, wbT, biasE, out);
}
